// SauvolaNet_11862699671641
// MI455X (gfx1250) — matrix operand order repaired and checked by reading, not compiled, not run on hardware
//
#include <hip/hip_runtime.h>
#include <hip/hip_bf16.h>

typedef __attribute__((ext_vector_type(16))) _Float16 v16h;
typedef __attribute__((ext_vector_type(8)))  float    v8f;
typedef __attribute__((ext_vector_type(4)))  unsigned int u32x4;
typedef __attribute__((ext_vector_type(8)))  int      i32x8;
typedef __attribute__((ext_vector_type(4)))  int      i32x4;

#define B_   4
#define H_   768
#define W_   768
#define HW_  (H_*W_)
#define PAD_ 10
#define HP_  (H_ + 2*PAD_)
#define WP_  (W_ + 2*PAD_)
#define NWIN 6
#define CW_  132            // 128-pixel strip + max halo (2*dil, dil<=2)

// A-matrix (16x32 f16) K mapping: lane half 0 -> K {0..7,16..23}, half 1 ->
// K {8..15,24..31}  (CDNA5 ISA 7.12.2, 16-bit A 16x32 layout)
__device__ __forceinline__ int kmapA(int half, int i) {
    return (i < 8) ? (half * 8 + i) : (16 + half * 8 + (i - 8));
}

// ---------------------------------------------------------------------------
// Implicit-GEMM conv3x3 (SAME, dilation d), fused InstanceNorm+ReLU on input.
// One block = one row, 128 consecutive pixels (8 waves x 16 pixels).
// Input halo tile staged in LDS (normalized f16); weights DMA'd to LDS by the
// Tensor Data Mover, then repacked into the WMMA B layout in LDS.
// ---------------------------------------------------------------------------
__global__ __launch_bounds__(256)
void conv_wmma_kernel(const float* __restrict__ in32,
                      const _Float16* __restrict__ in16,
                      const float* __restrict__ wgt,    // [Cout,Cin,3,3] f32
                      const float* __restrict__ bias,   // [Cout]
                      const float* __restrict__ stats,  // [B*Cin]{mean,rstd} or null
                      _Float16* __restrict__ out,       // [B,Cout,H,W] f16 (pre-norm)
                      int Cin, int Cout, int dil)
{
    __shared__ __attribute__((aligned(32))) _Float16 sIn[24 * 3 * CW_];  // 19.0 KB
    __shared__ __attribute__((aligned(16))) float    sWraw[4320];        // 17.3 KB
    __shared__ __attribute__((aligned(32))) _Float16 sB[12 * 512];       // 12.3 KB

    const int t    = threadIdx.x;
    const int lane = t & 31;
    const int wave = t >> 5;
    const int nseg = W_ / 128;                 // 6 strips per row
    int blk = blockIdx.x;                      // b*H*nseg + r*nseg + strip
    int b  = blk / (H_ * nseg);
    int rm = blk % (H_ * nseg);
    int r  = rm / nseg;
    int cbase = (rm % nseg) * 128;

    const int Ktot   = Cin * 9;
    const int nkc    = (Ktot + 31) >> 5;
    const int ntiles = (Cout + 15) >> 4;
    const int nW     = Cout * Ktot;            // #weight floats (<= 4320)

    // ---- TDM: raw weights -> LDS (1-D tile, data_size=4B) ----
    if (wave == 0) {
        unsigned long long ga = (unsigned long long)(uintptr_t)wgt;
        unsigned int ldsa = (unsigned int)(uintptr_t)(void*)sWraw; // low 32b = LDS offset
        u32x4 g0; i32x8 g1; i32x4 gz; i32x8 gz8;
        g0[0] = 1u;                                          // count=1, user desc
        g0[1] = ldsa;                                        // lds_addr
        g0[2] = (unsigned int)ga;                            // global_addr[31:0]
        g0[3] = (unsigned int)((ga >> 32) & 0x1FFFFFFu)      // global_addr[56:32]
              | (2u << 30);                                  // type = 2 ("image")
        g1[0] = 0x20000;                                     // data_size = 2 (4 bytes)
        g1[1] = (nW & 0xFFFF) << 16;                         // tensor_dim0[15:0]
        g1[2] = ((nW >> 16) & 0xFFFF) | (1 << 16);           // tensor_dim0[31:16], tensor_dim1=1
        g1[3] = (nW & 0xFFFF) << 16;                         // tile_dim0 = nW
        g1[4] = 1;                                           // tile_dim1 = 1
        g1[5] = nW;                                          // tensor_dim0_stride[31:0]
        g1[6] = 0;
        g1[7] = 0;
        gz[0] = gz[1] = gz[2] = gz[3] = 0;
        gz8[0] = gz8[1] = gz8[2] = gz8[3] = 0;
        gz8[4] = gz8[5] = gz8[6] = gz8[7] = 0;
        __builtin_amdgcn_tensor_load_to_lds(g0, g1, gz, gz, gz8, 0);
    }

    // ---- stage normalized input halo tile into LDS (overlaps TDM) ----
    const int cwid = 128 + 2 * dil;
    const int ntot = Cin * 3 * CW_;
    for (int idx = t; idx < ntot; idx += 256) {
        int cin = idx / (3 * CW_);
        int rme = idx % (3 * CW_);
        int kh  = rme / CW_;
        int j   = rme % CW_;
        float v = 0.f;
        int rr = r + (kh - 1) * dil;
        int cc = cbase + j - dil;
        if (j < cwid && rr >= 0 && rr < H_ && cc >= 0 && cc < W_) {
            size_t gidx = ((size_t)(b * Cin + cin)) * HW_ + (size_t)rr * W_ + cc;
            v = in32 ? in32[gidx] : (float)in16[gidx];
            if (stats) {
                float m  = stats[2 * (b * Cin + cin)];
                float rs = stats[2 * (b * Cin + cin) + 1];
                v = fmaxf((v - m) * rs, 0.f);
            }
        }
        sIn[idx] = (_Float16)v;
    }
    if (wave == 0) __builtin_amdgcn_s_wait_tensorcnt(0);
    __syncthreads();

    // ---- repack raw weights into WMMA B layout (32K x 16N per chunk) ----
    // B layout: lane half 0 holds K {0..7,16..23}, half 1 holds K {8..15,24..31}.
    // wraw flattening: w[n][cin][kh][kw] = sWraw[n*Ktot + K]  (K = cin*9+kh*3+kw)
    const int nbf = nkc * ntiles * 512;
    for (int idx = t; idx < nbf; idx += 256) {
        int i   = idx & 15;
        int ln  = (idx >> 4) & 31;
        int ntk = idx >> 9;                 // kc*ntiles + nt
        int kc  = ntk / ntiles;
        int nt  = ntk - kc * ntiles;
        int K = kc * 32 + kmapA(ln >> 4, i);
        int n = nt * 16 + (ln & 15);
        float v = 0.f;
        if (K < Ktot && n < Cout) v = sWraw[n * Ktot + K];
        sB[idx] = (_Float16)v;
    }
    __syncthreads();

    // ---- WMMA main loop: A from LDS input tile, B from LDS weight tiles ----
    const int half = lane >> 4;
    const int l16  = lane & 15;
    const int c0l  = wave * 16;             // local column base of this wave
    v8f acc0 = {};
    v8f acc1 = {};
    for (int kc = 0; kc < nkc; ++kc) {
        int kb = kc * 32;
        v16h a;
#pragma unroll
        for (int i = 0; i < 16; ++i) {
            int K = kb + kmapA(half, i);
            _Float16 v = (_Float16)0.f;
            if (K < Ktot) {
                int cin = K / 9;  int kk = K - cin * 9;
                int kh  = kk / 3; int kw = kk - kh * 3;
                v = sIn[(cin * 3 + kh) * CW_ + c0l + l16 + kw * dil];
            }
            a[i] = v;
        }
        v16h b0 = *(const v16h*)&sB[((kc * ntiles) * 32 + lane) * 16];
        acc0 = __builtin_amdgcn_wmma_f32_16x16x32_f16(false, a, false, b0,
                                                      (short)0, acc0, false, false);
        if (ntiles > 1) {
            v16h b1 = *(const v16h*)&sB[((kc * ntiles + 1) * 32 + lane) * 16];
            acc1 = __builtin_amdgcn_wmma_f32_16x16x32_f16(false, a, false, b1,
                                                          (short)0, acc1, false, false);
        }
    }

    // ---- store D (+bias) as f16 activations ----
#pragma unroll
    for (int v = 0; v < 8; ++v) {
        int pix = v + 8 * half;             // D: M = vgpr + 8*(lane/16)
        int col = cbase + c0l + pix;
        if (l16 < Cout)
            out[((size_t)(b * Cout + l16)) * HW_ + (size_t)r * W_ + col] =
                (_Float16)(acc0[v] + bias[l16]);
        if (ntiles > 1) {
            int n1 = 16 + l16;
            if (n1 < Cout)
                out[((size_t)(b * Cout + n1)) * HW_ + (size_t)r * W_ + col] =
                    (_Float16)(acc1[v] + bias[n1]);
        }
    }
}

// ---------------------------------------------------------------------------
// InstanceNorm statistics: one block per (b,c); mean + rstd over HxW.
// ---------------------------------------------------------------------------
__global__ __launch_bounds__(256)
void inorm_stats(const _Float16* __restrict__ act, float* __restrict__ stats)
{
    int bc = blockIdx.x;
    const _Float16* p = act + (size_t)bc * HW_;
    float s = 0.f, s2 = 0.f;
    for (int i = threadIdx.x; i < HW_; i += 256) {
        float v = (float)p[i];
        s += v; s2 += v * v;
    }
    __shared__ float r1[256], r2[256];
    int t = threadIdx.x;
    r1[t] = s; r2[t] = s2;
    __syncthreads();
    for (int off = 128; off > 0; off >>= 1) {
        if (t < off) { r1[t] += r1[t + off]; r2[t] += r2[t + off]; }
        __syncthreads();
    }
    if (t == 0) {
        float m   = r1[0] * (1.f / HW_);
        float var = r2[0] * (1.f / HW_) - m * m;
        stats[2 * bc]     = m;
        stats[2 * bc + 1] = rsqrtf(fmaxf(var, 0.f) + 1e-5f);
    }
}

// ---------------------------------------------------------------------------
// Softmax over the 6 window logits (channel dim).
// ---------------------------------------------------------------------------
__global__ __launch_bounds__(256)
void softmax6(const _Float16* __restrict__ logits, float* __restrict__ probs)
{
    int gid = blockIdx.x * 256 + threadIdx.x;
    if (gid >= B_ * HW_) return;
    int b = gid / HW_, pix = gid % HW_;
    size_t base = (size_t)b * NWIN * HW_ + pix;
    float l[NWIN], mx = -1e30f;
#pragma unroll
    for (int w = 0; w < NWIN; ++w) { l[w] = (float)logits[base + (size_t)w * HW_]; mx = fmaxf(mx, l[w]); }
    float s = 0.f;
#pragma unroll
    for (int w = 0; w < NWIN; ++w) { l[w] = expf(l[w] - mx); s += l[w]; }
    float inv = 1.f / s;
#pragma unroll
    for (int w = 0; w < NWIN; ++w) probs[base + (size_t)w * HW_] = l[w] * inv;
}

// ---------------------------------------------------------------------------
// Integral images of x and x^2 (padded by PAD_): row pass then column pass.
// ---------------------------------------------------------------------------
__global__ __launch_bounds__(256)
void integ_rows(const float* __restrict__ x, float* __restrict__ ii1, float* __restrict__ ii2)
{
    int blk = blockIdx.x;                 // b*HP_ + i
    int b = blk / HP_, i = blk % HP_;
    int t = threadIdx.x;
    int xr = i - PAD_;
    bool rowv = (xr >= 0 && xr < H_);
    const float* xrow = x + (size_t)b * HW_ + (size_t)(rowv ? xr : 0) * W_;
    float* o1 = ii1 + ((size_t)b * HP_ + i) * WP_;
    float* o2 = ii2 + ((size_t)b * HP_ + i) * WP_;
    __shared__ float s1[256], s2[256];
    float c1 = 0.f, c2 = 0.f;
    for (int j0 = 0; j0 < WP_; j0 += 256) {
        int j = j0 + t;
        float v = 0.f;
        if (rowv && j < WP_) {
            int xc = j - PAD_;
            if (xc >= 0 && xc < W_) v = xrow[xc];
        }
        s1[t] = v; s2[t] = v * v;
        __syncthreads();
        for (int off = 1; off < 256; off <<= 1) {
            float a1 = (t >= off) ? s1[t - off] : 0.f;
            float a2 = (t >= off) ? s2[t - off] : 0.f;
            __syncthreads();
            s1[t] += a1; s2[t] += a2;
            __syncthreads();
        }
        if (j < WP_) { o1[j] = c1 + s1[t]; o2[j] = c2 + s2[t]; }
        c1 += s1[255]; c2 += s2[255];
        __syncthreads();
    }
}

__global__ __launch_bounds__(256)
void integ_cols(float* __restrict__ ii1, float* __restrict__ ii2)
{
    int blk = blockIdx.x;                 // b*WP_ + j
    int b = blk / WP_, j = blk % WP_;
    int t = threadIdx.x;
    float* p1 = ii1 + (size_t)b * HP_ * WP_ + j;
    float* p2 = ii2 + (size_t)b * HP_ * WP_ + j;
    __shared__ float s1[256], s2[256];
    float c1 = 0.f, c2 = 0.f;
    for (int i0 = 0; i0 < HP_; i0 += 256) {
        int i = i0 + t;
        float v1 = (i < HP_) ? p1[(size_t)i * WP_] : 0.f;
        float v2 = (i < HP_) ? p2[(size_t)i * WP_] : 0.f;
        s1[t] = v1; s2[t] = v2;
        __syncthreads();
        for (int off = 1; off < 256; off <<= 1) {
            float a1 = (t >= off) ? s1[t - off] : 0.f;
            float a2 = (t >= off) ? s2[t - off] : 0.f;
            __syncthreads();
            s1[t] += a1; s2[t] += a2;
            __syncthreads();
        }
        if (i < HP_) { p1[(size_t)i * WP_] = c1 + s1[t]; p2[(size_t)i * WP_] = c2 + s2[t]; }
        c1 += s1[255]; c2 += s2[255];
        __syncthreads();
    }
}

// ---------------------------------------------------------------------------
// Final: Sauvola thresholds from the integral images (counts analytic),
// softmax-weighted combine, output (x - th1) * alpha.
// ---------------------------------------------------------------------------
__global__ __launch_bounds__(256)
void sauvola_final(const float* __restrict__ x,
                   const float* __restrict__ probs,
                   const float* __restrict__ ii1,
                   const float* __restrict__ ii2,
                   const float* __restrict__ kk,
                   const float* __restrict__ RR,
                   const float* __restrict__ alpha,
                   float* __restrict__ outp)
{
    const int wins[NWIN] = {3, 5, 7, 11, 15, 19};
    int gid = blockIdx.x * 256 + threadIdx.x;
    if (gid >= B_ * HW_) return;
    int b = gid / HW_, pix = gid % HW_;
    int r = pix / W_, c = pix % W_;
    const float* p1 = ii1 + (size_t)b * HP_ * WP_;
    const float* p2 = ii2 + (size_t)b * HP_ * WP_;
    float xv = x[(size_t)b * HW_ + pix];
    float th1 = 0.f;
#pragma unroll
    for (int w = 0; w < NWIN; ++w) {
        int hw  = wins[w];
        int top = 9 - hw / 2 + r, bot = top + hw;
        int lef = 9 - hw / 2 + c, rig = lef + hw;
        float s1 = p1[(size_t)bot * WP_ + rig] + p1[(size_t)top * WP_ + lef]
                 - p1[(size_t)top * WP_ + rig] - p1[(size_t)bot * WP_ + lef];
        float s2 = p2[(size_t)bot * WP_ + rig] + p2[(size_t)top * WP_ + lef]
                 - p2[(size_t)top * WP_ + rig] - p2[(size_t)bot * WP_ + lef];
        auto rcl = [](int i) {
            float v = (float)(i - PAD_ + 1);
            return fminf(fmaxf(v, 0.f), (float)H_);
        };
        float cnt = (rcl(bot) - rcl(top)) * (rcl(rig) - rcl(lef));
        float inv = 1.f / cnt;
        float Ex = s1 * inv, Ex2 = s2 * inv;
        float dev = sqrtf(fmaxf(Ex2 - Ex * Ex, 1e-6f));
        float th = Ex * (1.f + kk[w] * (dev / RR[w] - 1.f));
        th1 += probs[(size_t)b * NWIN * HW_ + (size_t)w * HW_ + pix] * th;
    }
    outp[(size_t)b * HW_ + pix] = (xv - th1) * alpha[0];
}

// ---------------------------------------------------------------------------
extern "C" void kernel_launch(void* const* d_in, const int* in_sizes, int n_in,
                              void* d_out, int out_size, void* d_ws, size_t ws_size,
                              hipStream_t stream)
{
    (void)in_sizes; (void)n_in; (void)out_size; (void)ws_size;

    const float* x   = (const float*)d_in[0];
    const float* w1  = (const float*)d_in[1];   const float* b1  = (const float*)d_in[2];
    const float* cw1 = (const float*)d_in[3];   const float* cb1 = (const float*)d_in[4];
    const float* cw2 = (const float*)d_in[5];   const float* cb2 = (const float*)d_in[6];
    const float* cw3 = (const float*)d_in[7];   const float* cb3 = (const float*)d_in[8];
    const float* cw4 = (const float*)d_in[9];   const float* cb4 = (const float*)d_in[10];
    const float* cw5 = (const float*)d_in[11];  const float* cb5 = (const float*)d_in[12];
    const float* w2  = (const float*)d_in[13];  const float* b2  = (const float*)d_in[14];
    const float* kk  = (const float*)d_in[15];
    const float* RR  = (const float*)d_in[16];
    const float* al  = (const float*)d_in[17];

    char* base = (char*)d_ws;
    const size_t ACT = (size_t)B_ * 24 * HW_ * sizeof(_Float16);   // 113.2 MB
    const size_t LOG = (size_t)B_ * NWIN * HW_ * sizeof(_Float16); // 28.3 MB
    const size_t PRB = (size_t)B_ * NWIN * HW_ * sizeof(float);    // 56.6 MB
    const size_t IIB = (size_t)B_ * HP_ * WP_ * sizeof(float);     // 9.9 MB

    _Float16* actA  = (_Float16*)(base);
    _Float16* actB  = (_Float16*)(base + ACT);
    _Float16* lg    = (_Float16*)(base + 2 * ACT);
    float*    probs = (float*)(base + 2 * ACT + LOG);
    float*    iix   = (float*)(base + 2 * ACT + LOG + PRB);
    float*    iix2  = (float*)(base + 2 * ACT + LOG + PRB + IIB);
    float*    stats = (float*)(base + 2 * ACT + LOG + PRB + 2 * IIB);

    dim3 blk(256);
    const int convGrid = B_ * H_ * (W_ / 128);         // 18432 blocks (1 row-strip each)
    const int pixGrid  = (B_ * HW_ + 255) / 256;

    // conv stack with fused inorm+relu on inputs
    conv_wmma_kernel<<<convGrid, blk, 0, stream>>>(x, nullptr, w1, b1, nullptr, actA, 1, 4, 1);
    inorm_stats<<<B_ * 4,  blk, 0, stream>>>(actA, stats);
    conv_wmma_kernel<<<convGrid, blk, 0, stream>>>(nullptr, actA, cw1, cb1, stats, actB, 4, 8, 2);
    inorm_stats<<<B_ * 8,  blk, 0, stream>>>(actB, stats);
    conv_wmma_kernel<<<convGrid, blk, 0, stream>>>(nullptr, actB, cw2, cb2, stats, actA, 8, 12, 2);
    inorm_stats<<<B_ * 12, blk, 0, stream>>>(actA, stats);
    conv_wmma_kernel<<<convGrid, blk, 0, stream>>>(nullptr, actA, cw3, cb3, stats, actB, 12, 16, 2);
    inorm_stats<<<B_ * 16, blk, 0, stream>>>(actB, stats);
    conv_wmma_kernel<<<convGrid, blk, 0, stream>>>(nullptr, actB, cw4, cb4, stats, actA, 16, 20, 2);
    inorm_stats<<<B_ * 20, blk, 0, stream>>>(actA, stats);
    conv_wmma_kernel<<<convGrid, blk, 0, stream>>>(nullptr, actA, cw5, cb5, stats, actB, 20, 24, 2);
    inorm_stats<<<B_ * 24, blk, 0, stream>>>(actB, stats);
    conv_wmma_kernel<<<convGrid, blk, 0, stream>>>(nullptr, actB, w2, b2, stats, lg, 24, 6, 1);

    // attention weights
    softmax6<<<pixGrid, blk, 0, stream>>>(lg, probs);

    // integral images of x and x^2
    integ_rows<<<B_ * HP_, blk, 0, stream>>>(x, iix, iix2);
    integ_cols<<<B_ * WP_, blk, 0, stream>>>(iix, iix2);

    // Sauvola thresholds + combine
    sauvola_final<<<pixGrid, blk, 0, stream>>>(x, probs, iix, iix2, kk, RR, al, (float*)d_out);
}
